// RNet_6725918785799
// MI455X (gfx1250) — compile-verified, matrix operand order checked by reading, not run on hardware
//
#include <hip/hip_runtime.h>

// ---------------------------------------------------------------------------
// R-Net forward for MI455X (gfx1250, wave32, WMMA).
// GEMMs use v_wmma_f32_16x16x32_f16 (f16 in, f32 accum). Carry-independent
// GEMMs are hoisted into parallel kernels; scans run as persistent 1-WG
// kernels with big LDS residency (320KB/WGP). Weights converted once per
// launch into padded f16 (N%16==0, K%32==0) so fragment loads are 2x b128.
// ---------------------------------------------------------------------------

typedef __attribute__((ext_vector_type(16))) _Float16 v16h;
typedef __attribute__((ext_vector_type(8)))  _Float16 v8h;
typedef __attribute__((ext_vector_type(8)))  float    v8f;

#define DEVINL __device__ __forceinline__

DEVINL float fsig(float x)  { return 1.0f / (1.0f + __expf(-x)); }
DEVINL float ftanh(float x) { return 2.0f / (1.0f + __expf(-2.0f * x)) - 1.0f; }

__device__ inline void zero_smem(void* p, int bytes) {
  int* w = (int*)p;
  int n = bytes >> 2;
  for (int i = threadIdx.x; i < n; i += blockDim.x) w[i] = 0;
}

__device__ inline void softmax_row(float* s, int n) {
  float mx = s[0];
  for (int i = 1; i < n; ++i) mx = fmaxf(mx, s[i]);
  float sum = 0.0f;
  for (int i = 0; i < n; ++i) { float e = __expf(s[i] - mx); s[i] = e; sum += e; }
  float inv = 1.0f / sum;
  for (int i = 0; i < n; ++i) s[i] *= inv;
}

// Load a 16x32 f16 fragment (A or B) from a row-major matrix.
// ISA layout: lane&15 -> row, lane>>4 selects which 8-half run; elements
// 0..7 = K[8*half .. +7], 8..15 = K[16+8*half .. +7].
DEVINL v16h ldfrag(const _Float16* p, int ld) {
  const int lane = threadIdx.x & 31;
  const _Float16* q = p + (lane & 15) * ld + ((lane >> 4) << 3);
  v8h lo = *(const v8h*)q;
  v8h hi = *(const v8h*)(q + 16);
  v16h o;
#pragma unroll
  for (int i = 0; i < 8; ++i) { o[i] = lo[i]; o[i + 8] = hi[i]; }
  return o;
}

// Workgroup-cooperative GEMM: C(32 x 16*Ntiles) = A(32 x 32*Kiters) @ W^T,
// W row-major (N x K) padded. Epilogue callback receives (m, n, acc).
template <class EPI>
DEVINL void wg_gemm(const _Float16* __restrict__ A, int lda,
                    const _Float16* __restrict__ W, int ldw,
                    int Ntiles, int Kiters, EPI epi) {
  const int lane = threadIdx.x & 31;
  const int wave = threadIdx.x >> 5;
  const int nw   = blockDim.x >> 5;
  const int nl   = lane & 15;
  const int mb0  = (lane >> 4) << 3;
  for (int t = wave; t < 2 * Ntiles; t += nw) {
    const int mt = t & 1, nt = t >> 1;
    const _Float16* arow = A + mt * 16 * lda;
    const _Float16* wrow = W + (size_t)nt * 16 * ldw;
    v8f acc = {};
    for (int k = 0; k < Kiters; ++k) {
      if (k + 1 < Kiters) __builtin_prefetch(wrow + nl * ldw + (k + 1) * 32, 0, 1);
      v16h a = ldfrag(arow + k * 32, lda);
      v16h b = ldfrag(wrow + k * 32, ldw);
      acc = __builtin_amdgcn_wmma_f32_16x16x32_f16(false, a, false, b, (short)0, acc,
                                                   false, false);
    }
    const int n  = nt * 16 + nl;
    const int mb = mt * 16 + mb0;
#pragma unroll
    for (int j = 0; j < 8; ++j) epi(mb + j, n, acc[j]);
  }
}

// ---------------------------------------------------------------------------
// Weight conversion f32 -> padded f16 (zero pads so WMMA pads contribute 0)
// ---------------------------------------------------------------------------
__global__ void k_convert(const float* __restrict__ src, int N, int K, int ld,
                          _Float16* __restrict__ dst, int Kp, int total) {
  int i = blockIdx.x * blockDim.x + threadIdx.x;
  if (i >= total) return;
  int n = i / Kp, k = i - n * Kp;
  float v = (n < N && k < K) ? src[(size_t)n * ld + k] : 0.0f;
  dst[i] = (_Float16)v;
}

// ---------------------------------------------------------------------------
// Encoder layer-0 input GEMM precompute (carry-independent, fully parallel)
// X0[t][cell][b][0..224] = emb(t) @ wih0[cell].T
// ---------------------------------------------------------------------------
__global__ void k_emb_gi(const float* __restrict__ Pw, const float* __restrict__ Pc,
                         const float* __restrict__ Qw, const float* __restrict__ Qc,
                         const _Float16* __restrict__ wih0,
                         float* __restrict__ X0P, float* __restrict__ X0Q) {
  extern __shared__ char smem[];
  _Float16* A = (_Float16*)smem;  // 32 x 512
  zero_smem(smem, 32 * 512 * 2);
  __syncthreads();
  int t = blockIdx.x;
  const float* srcW; const float* srcC; float* X0; int tt;
  if (t < 120) { srcW = Pw; srcC = Pc; X0 = X0P; tt = t; }
  else         { srcW = Qw; srcC = Qc; X0 = X0Q; tt = t - 120; }
  for (int idx = threadIdx.x; idx < 24 * 500; idx += blockDim.x) {
    int b = idx / 500, k = idx - b * 500;
    float v = (k < 300) ? srcW[((size_t)tt * 24 + b) * 300 + k]
                        : srcC[((size_t)tt * 24 + b) * 200 + (k - 300)];
    A[b * 512 + k] = (_Float16)v;
  }
  __syncthreads();
  for (int c = 0; c < 2; ++c) {
    float* dst = X0 + (size_t)(tt * 2 + c) * 24 * 225;
    wg_gemm(A, 512, wih0 + c * (240 * 512), 512, 15, 16,
            [&](int m, int n, float v) { if (m < 24 && n < 225) dst[(size_t)m * 225 + n] = v; });
  }
}

// ---------------------------------------------------------------------------
// Persistent 3-layer GRU encoder: block 0 = P (120 steps), block 1 = Q (32)
// ---------------------------------------------------------------------------
__global__ void k_encoder(const float* __restrict__ X0P, const float* __restrict__ X0Q,
                          const float* __restrict__ h0p, const float* __restrict__ h0q,
                          const _Float16* __restrict__ wih12, const _Float16* __restrict__ whh,
                          const float* __restrict__ bih, const float* __restrict__ bhh,
                          float* __restrict__ Up, float* __restrict__ Uq) {
  extern __shared__ char smem[];
  char* sp = smem;
  _Float16* hS   = (_Float16*)sp; sp += 6 * 32 * 96 * 2;
  _Float16* outA = (_Float16*)sp; sp += 32 * 160 * 2;
  _Float16* outB = (_Float16*)sp; sp += 32 * 160 * 2;
  float* g0 = (float*)sp; sp += 32 * 240 * 4;
  float* g1 = (float*)sp; sp += 32 * 240 * 4;
  float* g2 = (float*)sp; sp += 32 * 240 * 4;
  float* g3 = (float*)sp; sp += 32 * 240 * 4;
  float* hF = (float*)sp; sp += 6 * 1800 * 4;
  zero_smem(smem, (int)(sp - smem));
  __syncthreads();
  const int isQ = blockIdx.x;
  const int L = isQ ? 32 : 120;
  const float* X0 = isQ ? X0Q : X0P;
  const float* h0 = isQ ? h0q : h0p;
  float* U = isQ ? Uq : Up;
  for (int idx = threadIdx.x; idx < 6 * 1800; idx += blockDim.x) {
    int c = idx / 1800, r = idx - c * 1800, b = r / 75, u = r - b * 75;
    hF[idx] = h0[idx];
    hS[c * 3072 + b * 96 + u] = (_Float16)h0[idx];
  }
  __syncthreads();
  for (int t = 0; t < L; ++t) {
    // layer 0: recurrent GEMMs (input GEMMs precomputed in X0)
    wg_gemm(hS + 0 * 3072, 96, whh + 0 * (240 * 96), 96, 15, 3,
            [&](int m, int n, float v) { g2[m * 240 + n] = v; });
    wg_gemm(hS + 1 * 3072, 96, whh + 1 * (240 * 96), 96, 15, 3,
            [&](int m, int n, float v) { g3[m * 240 + n] = v; });
    __syncthreads();
    for (int idx = threadIdx.x; idx < 2 * 1800; idx += blockDim.x) {
      int c = idx / 1800, r = idx - c * 1800, b = r / 75, u = r - b * 75;
      const float* gi = X0 + ((size_t)(t * 2 + c) * 24 + b) * 225;
      const float* gh = (c ? g3 : g2) + b * 240;
      const float* bi = bih + c * 225; const float* bh = bhh + c * 225;
      float ir = gi[u] + bi[u], iz = gi[u + 75] + bi[u + 75], in_ = gi[u + 150] + bi[u + 150];
      float hr = gh[u] + bh[u], hz = gh[u + 75] + bh[u + 75], hn = gh[u + 150] + bh[u + 150];
      float rr = fsig(ir + hr), zz = fsig(iz + hz), nn = ftanh(in_ + rr * hn);
      float hv = (1.0f - zz) * nn + zz * hF[c * 1800 + b * 75 + u];
      hF[c * 1800 + b * 75 + u] = hv;
      hS[c * 3072 + b * 96 + u] = (_Float16)hv;
      outA[b * 160 + c * 75 + u] = (_Float16)hv;
      U[((size_t)t * 24 + b) * 450 + c * 75 + u] = hv;
    }
    __syncthreads();
    _Float16* oP = outA; _Float16* oN = outB;
    for (int layer = 1; layer <= 2; ++layer) {
      for (int c = 0; c < 2; ++c) {
        int ci = 2 * layer + c;
        wg_gemm(oP, 160, wih12 + (ci - 2) * (240 * 160), 160, 15, 5,
                [&](int m, int n, float v) { (c ? g1 : g0)[m * 240 + n] = v; });
        wg_gemm(hS + ci * 3072, 96, whh + ci * (240 * 96), 96, 15, 3,
                [&](int m, int n, float v) { (c ? g3 : g2)[m * 240 + n] = v; });
      }
      __syncthreads();
      for (int idx = threadIdx.x; idx < 2 * 1800; idx += blockDim.x) {
        int c = idx / 1800, r = idx - c * 1800, b = r / 75, u = r - b * 75;
        int ci = 2 * layer + c;
        const float* gi = (c ? g1 : g0) + b * 240;
        const float* gh = (c ? g3 : g2) + b * 240;
        const float* bi = bih + ci * 225; const float* bh = bhh + ci * 225;
        float ir = gi[u] + bi[u], iz = gi[u + 75] + bi[u + 75], in_ = gi[u + 150] + bi[u + 150];
        float hr = gh[u] + bh[u], hz = gh[u + 75] + bh[u + 75], hn = gh[u + 150] + bh[u + 150];
        float rr = fsig(ir + hr), zz = fsig(iz + hz), nn = ftanh(in_ + rr * hn);
        float hv = (1.0f - zz) * nn + zz * hF[ci * 1800 + b * 75 + u];
        hF[ci * 1800 + b * 75 + u] = hv;
        hS[ci * 3072 + b * 96 + u] = (_Float16)hv;
        oN[b * 160 + c * 75 + u] = (_Float16)hv;
        U[((size_t)t * 24 + b) * 450 + ci * 75 + u] = hv;
      }
      __syncthreads();
      _Float16* tmp = oP; oP = oN; oN = tmp;
    }
  }
}

// ---------------------------------------------------------------------------
// Parallel GEMM, A = [U[t] | U[t]] duplicated (Up2/Uq2 rows), per-block t.
// Used for Wuq, Xp = up@Wp.T and Gpre = up@WgA.T hoists.
// ---------------------------------------------------------------------------
__global__ void k_dup_gemm(const float* __restrict__ U, const _Float16* __restrict__ W,
                           int Ntiles, int Nreal, float* __restrict__ out) {
  extern __shared__ char smem[];
  _Float16* A = (_Float16*)smem;  // 32 x 928
  zero_smem(smem, 32 * 928 * 2);
  __syncthreads();
  int t = blockIdx.x;
  for (int idx = threadIdx.x; idx < 24 * 450; idx += blockDim.x) {
    int b = idx / 450, h = idx - b * 450;
    float v = U[((size_t)t * 24 + b) * 450 + h];
    A[b * 928 + h] = (_Float16)v;
    A[b * 928 + 450 + h] = (_Float16)v;
  }
  __syncthreads();
  wg_gemm(A, 928, W, 928, Ntiles, 29, [&](int m, int n, float v) {
    if (m < 24 && n < Nreal) out[((size_t)t * 24 + m) * Nreal + n] = v;
  });
}

// ---------------------------------------------------------------------------
// PQ matcher scan (120 steps), persistent single workgroup, ~293KB LDS.
// ---------------------------------------------------------------------------
__global__ void k_pq_seq(const float* __restrict__ XpP, const float* __restrict__ Wuq,
                         const float* __restrict__ Gpre, const float* __restrict__ UqF,
                         const float* __restrict__ pqV, const float* __restrict__ v0,
                         const _Float16* __restrict__ Wv, const _Float16* __restrict__ WgB,
                         const _Float16* __restrict__ gwih, const _Float16* __restrict__ gwhh,
                         const float* __restrict__ bih, const float* __restrict__ bhh,
                         float* __restrict__ vOut) {
  extern __shared__ char smem[];
  char* sp = smem;
  _Float16* vh = (_Float16*)sp; sp += 32 * 160 * 2;
  _Float16* cb = (_Float16*)sp; sp += 32 * 928 * 2;
  _Float16* cg = (_Float16*)sp; sp += 32 * 928 * 2;
  float* vf = (float*)sp; sp += 3600 * 4;
  float* xv = (float*)sp; sp += 32 * 160 * 4;
  float* Vv = (float*)sp; sp += 3600 * 4;
  float* sS = (float*)sp; sp += 24 * 32 * 4;
  float* gi = (float*)sp; sp += 32 * 464 * 4;
  float* gh = (float*)sp; sp += 32 * 464 * 4;
  zero_smem(smem, (int)(sp - smem));
  __syncthreads();
  for (int idx = threadIdx.x; idx < 3600; idx += blockDim.x) {
    int b = idx / 150, u = idx - b * 150;
    vf[idx] = v0[idx];
    Vv[idx] = pqV[idx];
    vh[b * 160 + u] = (_Float16)v0[idx];
  }
  __syncthreads();
  for (int i = 0; i < 120; ++i) {
    wg_gemm(vh, 160, Wv, 160, 10, 5, [&](int m, int n, float v) { xv[m * 160 + n] = v; });
    __syncthreads();
    // attention scores over Q (32) per batch
    for (int idx = threadIdx.x; idx < 768; idx += blockDim.x) {
      int b = idx % 24, j = idx / 24;
      const float* xp  = XpP + ((size_t)i * 24 + b) * 150;
      const float* wq  = Wuq + ((size_t)j * 24 + b) * 150;
      const float* xvb = xv + b * 160;
      const float* vvb = Vv + b * 150;
      float acc = 0.0f;
      for (int h = 0; h < 150; ++h) acc += ftanh(xp[h] + wq[h] + xvb[h]) * vvb[h];
      sS[b * 32 + j] = acc;
    }
    __syncthreads();
    if (threadIdx.x < 24) softmax_row(sS + threadIdx.x * 32, 32);
    __syncthreads();
    // context c (duplicated halves, as Uq2)
    for (int idx = threadIdx.x; idx < 24 * 450; idx += blockDim.x) {
      int b = idx / 450, h = idx - b * 450;
      float acc = 0.0f;
      for (int j = 0; j < 32; ++j) acc += sS[b * 32 + j] * UqF[((size_t)j * 24 + b) * 450 + h];
      cb[b * 928 + h] = (_Float16)acc;
      cb[b * 928 + 450 + h] = (_Float16)acc;
    }
    __syncthreads();
    // gate: only rows 900: of Wg matter; upi half hoisted into Gpre
    {
      const float* gp = Gpre + (size_t)i * 24 * 900;
      wg_gemm(cb, 928, WgB, 928, 57, 29, [&](int m, int n, float v) {
        if (m < 24 && n < 900) {
          float g = fsig(v + gp[(size_t)m * 900 + n]);
          cg[m * 928 + n] = (_Float16)(g * (float)cb[m * 928 + n]);
        }
      });
    }
    __syncthreads();
    wg_gemm(cg, 928, gwih, 928, 29, 29, [&](int m, int n, float v) { gi[m * 464 + n] = v; });
    wg_gemm(vh, 160, gwhh, 160, 29, 5,  [&](int m, int n, float v) { gh[m * 464 + n] = v; });
    __syncthreads();
    for (int idx = threadIdx.x; idx < 3600; idx += blockDim.x) {
      int b = idx / 150, u = idx - b * 150;
      float ir = gi[b * 464 + u] + bih[u], iz = gi[b * 464 + 150 + u] + bih[150 + u],
            in_ = gi[b * 464 + 300 + u] + bih[300 + u];
      float hr = gh[b * 464 + u] + bhh[u], hz = gh[b * 464 + 150 + u] + bhh[150 + u],
            hn = gh[b * 464 + 300 + u] + bhh[300 + u];
      float rr = fsig(ir + hr), zz = fsig(iz + hz), nn = ftanh(in_ + rr * hn);
      float nv = (1.0f - zz) * nn + zz * vf[idx];
      vf[idx] = nv;
      vh[b * 160 + u] = (_Float16)nv;
      vOut[((size_t)i * 24 + b) * 150 + u] = nv;
    }
    __syncthreads();
  }
}

// ---------------------------------------------------------------------------
// Self-matcher hoists: per-row GEMMs v@Wp.T and v@Wp2.T (parallel over 120)
// ---------------------------------------------------------------------------
__global__ void k_vrow_gemm2(const float* __restrict__ v,
                             const _Float16* __restrict__ W1, const _Float16* __restrict__ W2,
                             float* __restrict__ o1, float* __restrict__ o2) {
  extern __shared__ char smem[];
  _Float16* A = (_Float16*)smem;  // 32 x 160
  zero_smem(smem, 32 * 160 * 2);
  __syncthreads();
  int t = blockIdx.x;
  for (int idx = threadIdx.x; idx < 3600; idx += blockDim.x) {
    int b = idx / 150, h = idx - b * 150;
    A[b * 160 + h] = (_Float16)v[((size_t)t * 24 + b) * 150 + h];
  }
  __syncthreads();
  wg_gemm(A, 160, W1, 160, 10, 5, [&](int m, int n, float v_) {
    if (m < 24 && n < 150) o1[((size_t)t * 24 + m) * 150 + n] = v_; });
  wg_gemm(A, 160, W2, 160, 10, 5, [&](int m, int n, float v_) {
    if (m < 24 && n < 150) o2[((size_t)t * 24 + m) * 150 + n] = v_; });
}

// Self-matcher attention is carry-independent -> fully parallel over steps i.
__global__ void k_sm_att(const float* __restrict__ v, const float* __restrict__ Xsm,
                         const float* __restrict__ Wpv, const float* __restrict__ smV,
                         float* __restrict__ Csm) {
  extern __shared__ char smem[];
  float* sX = (float*)smem;   // 3600
  float* sV = sX + 3600;      // 3600
  float* sS = sV + 3600;      // 2880
  int i = blockIdx.x;
  for (int idx = threadIdx.x; idx < 3600; idx += blockDim.x) {
    sX[idx] = Xsm[(size_t)i * 3600 + idx];
    sV[idx] = smV[idx];
  }
  __syncthreads();
  for (int idx = threadIdx.x; idx < 2880; idx += blockDim.x) {
    int l = idx / 24, b = idx - l * 24;
    const float* wp = Wpv + ((size_t)l * 24 + b) * 150;
    const float* xb = sX + b * 150;
    const float* vb = sV + b * 150;
    float acc = 0.0f;
    for (int h = 0; h < 150; ++h) acc += ftanh(xb[h] + wp[h]) * vb[h];
    sS[b * 120 + l] = acc;
  }
  __syncthreads();
  if (threadIdx.x < 24) softmax_row(sS + threadIdx.x * 120, 120);
  __syncthreads();
  for (int idx = threadIdx.x; idx < 3600; idx += blockDim.x) {
    int b = idx / 150, h = idx - b * 150;
    float acc = 0.0f;
    for (int l = 0; l < 120; ++l) acc += sS[b * 120 + l] * v[((size_t)l * 24 + b) * 150 + h];
    Csm[((size_t)i * 24 + b) * 150 + h] = acc;
  }
}

// Self-matcher scan: 120 tiny GRU cells (attention already precomputed).
__global__ void k_sm_seq(const float* __restrict__ Csm, const float* __restrict__ h0,
                         const _Float16* __restrict__ gwih, const _Float16* __restrict__ gwhh,
                         const float* __restrict__ bih, const float* __restrict__ bhh,
                         float* __restrict__ hsOut) {
  extern __shared__ char smem[];
  char* sp = smem;
  _Float16* cbuf = (_Float16*)sp; sp += 32 * 160 * 2;
  _Float16* hb   = (_Float16*)sp; sp += 32 * 160 * 2;
  float* hf = (float*)sp; sp += 3600 * 4;
  float* gi = (float*)sp; sp += 32 * 464 * 4;
  float* gh = (float*)sp; sp += 32 * 464 * 4;
  zero_smem(smem, (int)(sp - smem));
  __syncthreads();
  for (int idx = threadIdx.x; idx < 3600; idx += blockDim.x) {
    int b = idx / 150, u = idx - b * 150;
    hf[idx] = h0[idx];
    hb[b * 160 + u] = (_Float16)h0[idx];
  }
  __syncthreads();
  for (int i = 0; i < 120; ++i) {
    for (int idx = threadIdx.x; idx < 3600; idx += blockDim.x) {
      int b = idx / 150, u = idx - b * 150;
      cbuf[b * 160 + u] = (_Float16)Csm[(size_t)i * 3600 + idx];
    }
    __syncthreads();
    wg_gemm(cbuf, 160, gwih, 160, 29, 5, [&](int m, int n, float v) { gi[m * 464 + n] = v; });
    wg_gemm(hb,   160, gwhh, 160, 29, 5, [&](int m, int n, float v) { gh[m * 464 + n] = v; });
    __syncthreads();
    for (int idx = threadIdx.x; idx < 3600; idx += blockDim.x) {
      int b = idx / 150, u = idx - b * 150;
      float ir = gi[b * 464 + u] + bih[u], iz = gi[b * 464 + 150 + u] + bih[150 + u],
            in_ = gi[b * 464 + 300 + u] + bih[300 + u];
      float hr = gh[b * 464 + u] + bhh[u], hz = gh[b * 464 + 150 + u] + bhh[150 + u],
            hn = gh[b * 464 + 300 + u] + bhh[300 + u];
      float rr = fsig(ir + hr), zz = fsig(iz + hz), nn = ftanh(in_ + rr * hn);
      float hv = (1.0f - zz) * nn + zz * hf[idx];
      hf[idx] = hv;
      hb[b * 160 + u] = (_Float16)hv;
      hsOut[(size_t)i * 3600 + idx] = hv;
    }
    __syncthreads();
  }
}

// ---------------------------------------------------------------------------
// Pointer network (single workgroup, one pass)
// ---------------------------------------------------------------------------
__global__ void k_pointer(const float* __restrict__ UqF, const float* __restrict__ hs,
                          const _Float16* __restrict__ WuW, const float* __restrict__ Wub,
                          const _Float16* __restrict__ WhF, const _Float16* __restrict__ WhaF,
                          const _Float16* __restrict__ gwih, const _Float16* __restrict__ gwhh,
                          const float* __restrict__ bih, const float* __restrict__ bhh,
                          const float* __restrict__ ptv,
                          float* __restrict__ WhhG, float* __restrict__ giG,
                          float* __restrict__ ghG, float* __restrict__ out) {
  extern __shared__ char smem[];
  char* sp = smem;
  float* Vv = (float*)sp; sp += 10800 * 4;
  float* rf = (float*)sp; sp += 10800 * 4;
  float* xb = (float*)sp; sp += 32 * 464 * 4;
  float* ra = (float*)sp; sp += 32 * 464 * 4;
  float* sS = (float*)sp; sp += 24 * 32 * 4;
  float* s2 = (float*)sp; sp += 24 * 120 * 4;
  _Float16* Ab = (_Float16*)sp; sp += 32 * 480 * 2;
  _Float16* rh = (_Float16*)sp; sp += 32 * 480 * 2;
  zero_smem(smem, (int)(sp - smem));
  __syncthreads();
  for (int idx = threadIdx.x; idx < 10800; idx += blockDim.x) Vv[idx] = ptv[idx];
  __syncthreads();
  // --- attention over Q: x = tanh(Uq@Wu.T + b), s = <x, Vv>
  for (int j = 0; j < 32; ++j) {
    for (int idx = threadIdx.x; idx < 10800; idx += blockDim.x) {
      int b = idx / 450, h = idx - b * 450;
      Ab[b * 480 + h] = (_Float16)UqF[(size_t)j * 10800 + idx];
    }
    __syncthreads();
    wg_gemm(Ab, 480, WuW, 480, 29, 15, [&](int m, int n, float v) {
      if (n < 450) xb[m * 464 + n] = ftanh(v + Wub[n]); });
    __syncthreads();
    if (threadIdx.x < 24) {
      int b = threadIdx.x; float acc = 0.0f;
      for (int h = 0; h < 450; ++h) acc += xb[b * 464 + h] * Vv[b * 450 + h];
      sS[b * 32 + j] = acc;
    }
    __syncthreads();
  }
  if (threadIdx.x < 24) softmax_row(sS + threadIdx.x * 32, 32);
  __syncthreads();
  for (int idx = threadIdx.x; idx < 10800; idx += blockDim.x) {
    int b = idx / 450, h = idx - b * 450;
    float acc = 0.0f;
    for (int j = 0; j < 32; ++j) acc += sS[b * 32 + j] * UqF[(size_t)j * 10800 + b * 450 + h];
    rf[idx] = acc;
    rh[b * 480 + h] = (_Float16)acc;
  }
  __syncthreads();
  // --- Wh_h = hs @ Wh.T for all 120 positions
  for (int l = 0; l < 120; ++l) {
    for (int idx = threadIdx.x; idx < 24 * 160; idx += blockDim.x) {
      int b = idx / 160, u = idx - b * 160;
      Ab[b * 160 + u] = (u < 150) ? (_Float16)hs[(size_t)l * 3600 + b * 150 + u] : (_Float16)0.0f;
    }
    __syncthreads();
    wg_gemm(Ab, 160, WhF, 160, 29, 5, [&](int m, int n, float v) {
      if (m < 24 && n < 450) WhhG[(size_t)l * 10800 + m * 450 + n] = v; });
    __syncthreads();
  }
  // --- p1
  wg_gemm(rh, 480, WhaF, 480, 29, 15, [&](int m, int n, float v) { ra[m * 464 + n] = v; });
  __syncthreads();
  for (int idx = threadIdx.x; idx < 2880; idx += blockDim.x) {
    int l = idx / 24, b = idx - l * 24;
    const float* wh = WhhG + (size_t)l * 10800 + b * 450;
    float acc = 0.0f;
    for (int h = 0; h < 450; ++h) acc += ftanh(wh[h] + ra[b * 464 + h]) * Vv[b * 450 + h];
    s2[b * 120 + l] = acc;
  }
  __syncthreads();
  if (threadIdx.x < 24) {
    int b = threadIdx.x;
    softmax_row(s2 + b * 120, 120);
    for (int l = 0; l < 120; ++l) out[b * 120 + l] = s2[b * 120 + l];
  }
  __syncthreads();
  // --- context over P and pointer GRU
  for (int idx = threadIdx.x; idx < 3600; idx += blockDim.x) {
    int b = idx / 150, h = idx - b * 150;
    float acc = 0.0f;
    for (int l = 0; l < 120; ++l) acc += s2[b * 120 + l] * hs[(size_t)l * 3600 + b * 150 + h];
    Ab[b * 160 + h] = (_Float16)acc;   // pads already zero from Wh_h fill
  }
  __syncthreads();
  wg_gemm(Ab, 160, gwih, 160, 85, 5, [&](int m, int n, float v) {
    if (m < 24 && n < 1350) giG[(size_t)m * 1350 + n] = v; });
  wg_gemm(rh, 480, gwhh, 480, 85, 15, [&](int m, int n, float v) {
    if (m < 24 && n < 1350) ghG[(size_t)m * 1350 + n] = v; });
  __syncthreads();
  for (int idx = threadIdx.x; idx < 10800; idx += blockDim.x) {
    int b = idx / 450, u = idx - b * 450;
    float ir = giG[(size_t)b * 1350 + u] + bih[u],
          iz = giG[(size_t)b * 1350 + 450 + u] + bih[450 + u],
          in_ = giG[(size_t)b * 1350 + 900 + u] + bih[900 + u];
    float hr = ghG[(size_t)b * 1350 + u] + bhh[u],
          hz = ghG[(size_t)b * 1350 + 450 + u] + bhh[450 + u],
          hn = ghG[(size_t)b * 1350 + 900 + u] + bhh[900 + u];
    float rr = fsig(ir + hr), zz = fsig(iz + hz), nn = ftanh(in_ + rr * hn);
    float nv = (1.0f - zz) * nn + zz * rf[idx];
    rh[b * 480 + u] = (_Float16)nv;
  }
  __syncthreads();
  // --- p2
  wg_gemm(rh, 480, WhaF, 480, 29, 15, [&](int m, int n, float v) { ra[m * 464 + n] = v; });
  __syncthreads();
  for (int idx = threadIdx.x; idx < 2880; idx += blockDim.x) {
    int l = idx / 24, b = idx - l * 24;
    const float* wh = WhhG + (size_t)l * 10800 + b * 450;
    float acc = 0.0f;
    for (int h = 0; h < 450; ++h) acc += ftanh(wh[h] + ra[b * 464 + h]) * Vv[b * 450 + h];
    s2[b * 120 + l] = acc;
  }
  __syncthreads();
  if (threadIdx.x < 24) {
    int b = threadIdx.x;
    softmax_row(s2 + b * 120, 120);
    for (int l = 0; l < 120; ++l) out[2880 + b * 120 + l] = s2[b * 120 + l];
  }
}

// ---------------------------------------------------------------------------
// Workspace layout (bytes, all chunks 64B-aligned by construction)
// ---------------------------------------------------------------------------
namespace ws_off {
constexpr size_t o_wih0  = 0;
constexpr size_t o_wih12 = o_wih0  + 2ull * 240 * 512 * 2;
constexpr size_t o_whh   = o_wih12 + 4ull * 240 * 160 * 2;
constexpr size_t o_Wq    = o_whh   + 6ull * 240 * 96 * 2;
constexpr size_t o_Wp    = o_Wq    + 160ull * 928 * 2;
constexpr size_t o_Wv    = o_Wp    + 160ull * 928 * 2;
constexpr size_t o_WgA   = o_Wv    + 160ull * 160 * 2;
constexpr size_t o_WgB   = o_WgA   + 912ull * 928 * 2;
constexpr size_t o_pqwih = o_WgB   + 912ull * 928 * 2;
constexpr size_t o_pqwhh = o_pqwih + 464ull * 928 * 2;
constexpr size_t o_smWp  = o_pqwhh + 464ull * 160 * 2;
constexpr size_t o_smWp2 = o_smWp  + 160ull * 160 * 2;
constexpr size_t o_smwih = o_smWp2 + 160ull * 160 * 2;
constexpr size_t o_smwhh = o_smwih + 464ull * 160 * 2;
constexpr size_t o_WuW   = o_smwhh + 464ull * 160 * 2;
constexpr size_t o_Wh    = o_WuW   + 464ull * 480 * 2;
constexpr size_t o_Wha   = o_Wh    + 464ull * 160 * 2;
constexpr size_t o_ptwih = o_Wha   + 464ull * 480 * 2;
constexpr size_t o_ptwhh = o_ptwih + 1360ull * 160 * 2;
constexpr size_t o_X0P   = o_ptwhh + 1360ull * 480 * 2;
constexpr size_t o_X0Q   = o_X0P + 120ull * 2 * 24 * 225 * 4;
constexpr size_t o_Up    = o_X0Q + 32ull * 2 * 24 * 225 * 4;
constexpr size_t o_Uq    = o_Up  + 120ull * 24 * 450 * 4;
constexpr size_t o_Wuq   = o_Uq  + 32ull * 24 * 450 * 4;
constexpr size_t o_XpP   = o_Wuq + 32ull * 24 * 150 * 4;
constexpr size_t o_Gpre  = o_XpP + 120ull * 24 * 150 * 4;
constexpr size_t o_v     = o_Gpre + 120ull * 24 * 900 * 4;
constexpr size_t o_Xsm   = o_v   + 120ull * 24 * 150 * 4;
constexpr size_t o_Wpv   = o_Xsm + 120ull * 24 * 150 * 4;
constexpr size_t o_Csm   = o_Wpv + 120ull * 24 * 150 * 4;
constexpr size_t o_hs    = o_Csm + 120ull * 24 * 150 * 4;
constexpr size_t o_Whh   = o_hs  + 120ull * 24 * 150 * 4;
constexpr size_t o_giP   = o_Whh + 120ull * 24 * 450 * 4;
constexpr size_t o_ghP   = o_giP + 24ull * 1350 * 4;
}  // namespace ws_off

constexpr int SM_EMB  = 32 * 512 * 2;
constexpr int SM_ENC  = 6 * 32 * 96 * 2 + 2 * 32 * 160 * 2 + 4 * 32 * 240 * 4 + 6 * 1800 * 4;
constexpr int SM_DUP  = 32 * 928 * 2;
constexpr int SM_VROW = 32 * 160 * 2;
constexpr int SM_SMA  = (3600 + 3600 + 2880) * 4;
constexpr int SM_PQ   = 32 * 160 * 2 + 2 * 32 * 928 * 2 + 3600 * 4 + 32 * 160 * 4 + 3600 * 4 +
                        24 * 32 * 4 + 2 * 32 * 464 * 4;
constexpr int SM_SMS  = 2 * 32 * 160 * 2 + 3600 * 4 + 2 * 32 * 464 * 4;
constexpr int SM_PTR  = 2 * 10800 * 4 + 2 * 32 * 464 * 4 + 24 * 32 * 4 + 24 * 120 * 4 +
                        2 * 32 * 480 * 2;

extern "C" void kernel_launch(void* const* d_in, const int* in_sizes, int n_in,
                              void* d_out, int out_size, void* d_ws, size_t ws_size,
                              hipStream_t stream) {
  (void)in_sizes; (void)n_in; (void)out_size; (void)ws_size;
  using namespace ws_off;
  char* ws = (char*)d_ws;
  auto F16P = [&](size_t off) { return (_Float16*)(ws + off); };
  auto F32P = [&](size_t off) { return (float*)(ws + off); };

  const float* Pw        = (const float*)d_in[0];
  const float* Pc        = (const float*)d_in[1];
  const float* Qw        = (const float*)d_in[2];
  const float* Qc        = (const float*)d_in[3];
  const float* enc_wih0  = (const float*)d_in[4];
  const float* enc_wih12 = (const float*)d_in[5];
  const float* enc_whh   = (const float*)d_in[6];
  const float* enc_bih   = (const float*)d_in[7];
  const float* enc_bhh   = (const float*)d_in[8];
  const float* enc_h0p   = (const float*)d_in[9];
  const float* enc_h0q   = (const float*)d_in[10];
  const float* pq_Wp     = (const float*)d_in[11];
  const float* pq_Wq     = (const float*)d_in[12];
  const float* pq_Wv     = (const float*)d_in[13];
  const float* pq_Wg     = (const float*)d_in[14];
  const float* pq_gwih   = (const float*)d_in[15];
  const float* pq_gwhh   = (const float*)d_in[16];
  const float* pq_bih    = (const float*)d_in[17];
  const float* pq_bhh    = (const float*)d_in[18];
  const float* pq_v0     = (const float*)d_in[19];
  const float* pq_V      = (const float*)d_in[20];
  const float* sm_Wp     = (const float*)d_in[21];
  const float* sm_Wp2    = (const float*)d_in[22];
  const float* sm_gwih   = (const float*)d_in[23];
  const float* sm_gwhh   = (const float*)d_in[24];
  const float* sm_bih    = (const float*)d_in[25];
  const float* sm_bhh    = (const float*)d_in[26];
  const float* sm_h0     = (const float*)d_in[27];
  const float* sm_V      = (const float*)d_in[28];
  const float* pt_Wu_w   = (const float*)d_in[29];
  const float* pt_Wu_b   = (const float*)d_in[30];
  const float* pt_Wh     = (const float*)d_in[31];
  const float* pt_Wha    = (const float*)d_in[32];
  const float* pt_gwih   = (const float*)d_in[33];
  const float* pt_gwhh   = (const float*)d_in[34];
  const float* pt_bih    = (const float*)d_in[35];
  const float* pt_bhh    = (const float*)d_in[36];
  const float* pt_v      = (const float*)d_in[37];

  // allow >48KB dynamic LDS (WGP has 320KB)
  hipFuncSetAttribute((const void*)k_encoder, hipFuncAttributeMaxDynamicSharedMemorySize, SM_ENC);
  hipFuncSetAttribute((const void*)k_dup_gemm, hipFuncAttributeMaxDynamicSharedMemorySize, SM_DUP);
  hipFuncSetAttribute((const void*)k_pq_seq,  hipFuncAttributeMaxDynamicSharedMemorySize, SM_PQ);
  hipFuncSetAttribute((const void*)k_sm_seq,  hipFuncAttributeMaxDynamicSharedMemorySize, SM_SMS);
  hipFuncSetAttribute((const void*)k_pointer, hipFuncAttributeMaxDynamicSharedMemorySize, SM_PTR);

  auto conv = [&](const float* src, int N, int K, int ld, size_t off, int Np, int Kp) {
    int total = Np * Kp;
    k_convert<<<dim3((total + 255) / 256), dim3(256), 0, stream>>>(src, N, K, ld, F16P(off), Kp,
                                                                   total);
  };
  // --- weight conversion (fp32 -> padded f16)
  for (int c = 0; c < 2; ++c)
    conv(enc_wih0 + (size_t)c * 225 * 500, 225, 500, 500, o_wih0 + (size_t)c * 240 * 512 * 2, 240, 512);
  for (int m = 0; m < 4; ++m)
    conv(enc_wih12 + (size_t)m * 225 * 150, 225, 150, 150, o_wih12 + (size_t)m * 240 * 160 * 2, 240, 160);
  for (int m = 0; m < 6; ++m)
    conv(enc_whh + (size_t)m * 225 * 75, 225, 75, 75, o_whh + (size_t)m * 240 * 96 * 2, 240, 96);
  conv(pq_Wq, 150, 900, 900, o_Wq, 160, 928);
  conv(pq_Wp, 150, 900, 900, o_Wp, 160, 928);
  conv(pq_Wv, 150, 150, 150, o_Wv, 160, 160);
  conv(pq_Wg + 900ull * 1800, 900, 900, 1800, o_WgA, 912, 928);        // rows 900:, cols :900
  conv(pq_Wg + 900ull * 1800 + 900, 900, 900, 1800, o_WgB, 912, 928);  // rows 900:, cols 900:
  conv(pq_gwih, 450, 900, 900, o_pqwih, 464, 928);
  conv(pq_gwhh, 450, 150, 150, o_pqwhh, 464, 160);
  conv(sm_Wp, 150, 150, 150, o_smWp, 160, 160);
  conv(sm_Wp2, 150, 150, 150, o_smWp2, 160, 160);
  conv(sm_gwih, 450, 150, 150, o_smwih, 464, 160);
  conv(sm_gwhh, 450, 150, 150, o_smwhh, 464, 160);
  conv(pt_Wu_w, 450, 450, 450, o_WuW, 464, 480);
  conv(pt_Wh, 450, 150, 150, o_Wh, 464, 160);
  conv(pt_Wha, 450, 450, 450, o_Wha, 464, 480);
  conv(pt_gwih, 1350, 150, 150, o_ptwih, 1360, 160);
  conv(pt_gwhh, 1350, 450, 450, o_ptwhh, 1360, 480);

  // --- encoder
  k_emb_gi<<<dim3(152), dim3(512), SM_EMB, stream>>>(Pw, Pc, Qw, Qc, F16P(o_wih0), F32P(o_X0P),
                                                     F32P(o_X0Q));
  k_encoder<<<dim3(2), dim3(512), SM_ENC, stream>>>(F32P(o_X0P), F32P(o_X0Q), enc_h0p, enc_h0q,
                                                    F16P(o_wih12), F16P(o_whh), enc_bih, enc_bhh,
                                                    F32P(o_Up), F32P(o_Uq));
  // --- PQ matcher hoists + scan
  k_dup_gemm<<<dim3(32), dim3(512), SM_DUP, stream>>>(F32P(o_Uq), F16P(o_Wq), 10, 150,
                                                      F32P(o_Wuq));
  k_dup_gemm<<<dim3(120), dim3(512), SM_DUP, stream>>>(F32P(o_Up), F16P(o_Wp), 10, 150,
                                                       F32P(o_XpP));
  k_dup_gemm<<<dim3(120), dim3(512), SM_DUP, stream>>>(F32P(o_Up), F16P(o_WgA), 57, 900,
                                                       F32P(o_Gpre));
  k_pq_seq<<<dim3(1), dim3(512), SM_PQ, stream>>>(F32P(o_XpP), F32P(o_Wuq), F32P(o_Gpre),
                                                  F32P(o_Uq), pq_V, pq_v0, F16P(o_Wv),
                                                  F16P(o_WgB), F16P(o_pqwih), F16P(o_pqwhh),
                                                  pq_bih, pq_bhh, F32P(o_v));
  // --- self matcher hoists + scan
  k_vrow_gemm2<<<dim3(120), dim3(512), SM_VROW, stream>>>(F32P(o_v), F16P(o_smWp), F16P(o_smWp2),
                                                          F32P(o_Xsm), F32P(o_Wpv));
  k_sm_att<<<dim3(120), dim3(512), SM_SMA, stream>>>(F32P(o_v), F32P(o_Xsm), F32P(o_Wpv), sm_V,
                                                     F32P(o_Csm));
  k_sm_seq<<<dim3(1), dim3(512), SM_SMS, stream>>>(F32P(o_Csm), sm_h0, F16P(o_smwih),
                                                   F16P(o_smwhh), sm_bih, sm_bhh, F32P(o_hs));
  // --- pointer
  k_pointer<<<dim3(1), dim3(512), SM_PTR, stream>>>(F32P(o_Uq), F32P(o_hs), F16P(o_WuW), pt_Wu_b,
                                                    F16P(o_Wh), F16P(o_Wha), F16P(o_ptwih),
                                                    F16P(o_ptwhh), pt_bih, pt_bhh, pt_v,
                                                    F32P(o_Whh), F32P(o_giP), F32P(o_ghP),
                                                    (float*)d_out);
}
